// Agnn_33337536151793
// MI455X (gfx1250) — matrix operand order repaired and checked by reading, not compiled, not run on hardware
//
#include <hip/hip_runtime.h>
#include <hip/hip_bf16.h>

typedef __attribute__((ext_vector_type(16))) _Float16 v16h;
typedef __attribute__((ext_vector_type(8)))  float    v8f;

#define FIN 512
#define HD  16
#define NCLASS 40

// ---------------------------------------------------------------------------
// Kernel 1: h = relu(x @ W1 + b1), [N,512]x[512,16] -> [N,16]
// f16 WMMA 16x16x32, f32 accumulate. One wave = one 16-row tile of x.
// W1^T staged in LDS as f16 (16 KB), matching the dense 16-bit B layout.
// ---------------------------------------------------------------------------
__global__ __launch_bounds__(256) void gemm1_relu(
    const float* __restrict__ x, const float* __restrict__ W1,
    const float* __restrict__ b1, float* __restrict__ h, int N) {
  __shared__ alignas(32) _Float16 w1t[HD * FIN];   // [n][k], row = 1024 B
  for (int i = threadIdx.x; i < HD * FIN; i += 256) {
    int n = i >> 9;          // i / 512
    int k = i & (FIN - 1);   // i % 512
    w1t[i] = (_Float16)W1[k * HD + n];
  }
  __syncthreads();

  const int wave = threadIdx.x >> 5;
  const int lane = threadIdx.x & 31;
  const int half = lane >> 4;    // 0: K 0..15 of each 32-chunk; 1: K 16..31
  const int mrow = lane & 15;
  const int numTiles = (N + 15) >> 4;
  const int tile = blockIdx.x * 8 + wave;
  if (tile >= numTiles) return;            // wave-uniform: EXEC stays all-1s
  const int row0 = tile << 4;

  int rload = row0 + mrow;
  if (rload > N - 1) rload = N - 1;        // clamp keeps EXEC full in tail tile
  const float4* xr = (const float4*)(x + (size_t)rload * FIN);

  v8f c = {};
#pragma unroll
  for (int kc = 0; kc < FIN / 32; ++kc) {
    // A tile: row = mrow, K = kc*32 + half*16 + [0..16)
    float4 p0 = xr[kc * 8 + half * 4 + 0];
    float4 p1 = xr[kc * 8 + half * 4 + 1];
    float4 p2 = xr[kc * 8 + half * 4 + 2];
    float4 p3 = xr[kc * 8 + half * 4 + 3];
    v16h a;
    a[0]  = (_Float16)p0.x; a[1]  = (_Float16)p0.y;
    a[2]  = (_Float16)p0.z; a[3]  = (_Float16)p0.w;
    a[4]  = (_Float16)p1.x; a[5]  = (_Float16)p1.y;
    a[6]  = (_Float16)p1.z; a[7]  = (_Float16)p1.w;
    a[8]  = (_Float16)p2.x; a[9]  = (_Float16)p2.y;
    a[10] = (_Float16)p2.z; a[11] = (_Float16)p2.w;
    a[12] = (_Float16)p3.x; a[13] = (_Float16)p3.y;
    a[14] = (_Float16)p3.z; a[15] = (_Float16)p3.w;
    // B tile: N = mrow, K = kc*32 + half*16 + [0..16), contiguous f16 in LDS
    const v16h b = *(const v16h*)&w1t[(mrow << 9) + (kc << 5) + (half << 4)];
    c = __builtin_amdgcn_wmma_f32_16x16x32_f16(
        false, a, false, b, (short)0, c, false, false);
  }

  // C/D layout: VGPR v -> M = half*8 + v, N = mrow
  const float bias = b1[mrow];
#pragma unroll
  for (int v = 0; v < 8; ++v) {
    int r = row0 + half * 8 + v;
    if (r < N) {
      float val = c[v] + bias;
      h[(size_t)r * HD + mrow] = val > 0.f ? val : 0.f;
    }
  }
}

// ---------------------------------------------------------------------------
// Edge-list compaction: int64 (2,E) + appended self-loops -> int32 src/dst
// ---------------------------------------------------------------------------
__global__ __launch_bounds__(256) void build_edges(
    const long long* __restrict__ ei, int* __restrict__ s32,
    int* __restrict__ d32, int E, int ET) {
  int e = blockIdx.x * 256 + threadIdx.x;
  if (e >= ET) return;
  if (e < E) {
    s32[e] = (int)ei[e];
    d32[e] = (int)ei[(size_t)E + e];
  } else {
    s32[e] = e - E;
    d32[e] = e - E;
  }
}

// ---------------------------------------------------------------------------
// Row L2-normalize (eps = 1e-12), 16-wide rows
// ---------------------------------------------------------------------------
__global__ __launch_bounds__(256) void rownorm16(
    const float* __restrict__ h, float* __restrict__ xn, int N) {
  int n = blockIdx.x * 256 + threadIdx.x;
  if (n >= N) return;
  const float4* p = (const float4*)(h + (size_t)n * HD);
  float4 a = p[0], b = p[1], c = p[2], d = p[3];
  float ss = a.x*a.x + a.y*a.y + a.z*a.z + a.w*a.w
           + b.x*b.x + b.y*b.y + b.z*b.z + b.w*b.w
           + c.x*c.x + c.y*c.y + c.z*c.z + c.w*c.w
           + d.x*d.x + d.y*d.y + d.z*d.z + d.w*d.w;
  float inv = 1.0f / fmaxf(sqrtf(ss), 1e-12f);
  float4* q = (float4*)(xn + (size_t)n * HD);
  a.x*=inv; a.y*=inv; a.z*=inv; a.w*=inv;
  b.x*=inv; b.y*=inv; b.z*=inv; b.w*=inv;
  c.x*=inv; c.y*=inv; c.z*=inv; c.w*=inv;
  d.x*=inv; d.y*=inv; d.z*=inv; d.w*=inv;
  q[0]=a; q[1]=b; q[2]=c; q[3]=d;
}

// ---------------------------------------------------------------------------
// Per-layer zero init: segment-max (flipped-uint -inf), segment-sum, out rows
// ---------------------------------------------------------------------------
__global__ __launch_bounds__(256) void seg_init(
    unsigned* __restrict__ m_u, float* __restrict__ ssum,
    float* __restrict__ hout, int N) {
  int n = blockIdx.x * 256 + threadIdx.x;
  if (n >= N) return;
  m_u[n] = 0u;          // below flip() of every finite float
  ssum[n] = 0.f;
  float4 z = {0.f, 0.f, 0.f, 0.f};
  float4* q = (float4*)(hout + (size_t)n * HD);
  q[0]=z; q[1]=z; q[2]=z; q[3]=z;
}

__device__ __forceinline__ unsigned flip_f32(float f) {
  unsigned u = __float_as_uint(f);
  return u ^ ((u & 0x80000000u) ? 0xFFFFFFFFu : 0x80000000u);
}
__device__ __forceinline__ float unflip_f32(unsigned u) {
  u ^= ((u & 0x80000000u) ? 0x80000000u : 0xFFFFFFFFu);
  return __uint_as_float(u);
}

// ---------------------------------------------------------------------------
// Pass 1: alpha[e] = beta * <xn[src], xn[dst]>; segment max via flipped uint
// (xn table is 6.4 MB -> L2-resident gathers)
// ---------------------------------------------------------------------------
__global__ __launch_bounds__(256) void edge_alpha(
    const float* __restrict__ xn, const int* __restrict__ s32,
    const int* __restrict__ d32, const float* __restrict__ beta,
    float* __restrict__ alpha, unsigned* __restrict__ m_u, int ET) {
  int e = blockIdx.x * 256 + threadIdx.x;
  if (e >= ET) return;
  int s = s32[e], d = d32[e];
  const float4* ps = (const float4*)(xn + (size_t)s * HD);
  const float4* pd = (const float4*)(xn + (size_t)d * HD);
  float acc = 0.f;
#pragma unroll
  for (int i = 0; i < 4; ++i) {
    float4 a = ps[i], b = pd[i];
    acc += a.x*b.x + a.y*b.y + a.z*b.z + a.w*b.w;
  }
  float al = beta[0] * acc;
  alpha[e] = al;
  atomicMax(m_u + d, flip_f32(al));
}

// ---------------------------------------------------------------------------
// Pass 2: e = exp(alpha - m[dst]) (in place); segment sum via f32 atomics
// ---------------------------------------------------------------------------
__global__ __launch_bounds__(256) void edge_exp(
    float* __restrict__ albuf, const int* __restrict__ d32,
    const unsigned* __restrict__ m_u, float* __restrict__ ssum, int ET) {
  int e = blockIdx.x * 256 + threadIdx.x;
  if (e >= ET) return;
  int d = d32[e];
  float m = unflip_f32(m_u[d]);
  float ev = expf(albuf[e] - m);
  albuf[e] = ev;
  atomicAdd(ssum + d, ev);
}

// ---------------------------------------------------------------------------
// Pass 3: out[dst] += (e / (s[dst]+1e-16)) * hin[src]; thread = (edge,feature)
// ---------------------------------------------------------------------------
__global__ __launch_bounds__(256) void edge_aggr(
    const float* __restrict__ ebuf, const int* __restrict__ s32,
    const int* __restrict__ d32, const float* __restrict__ ssum,
    const float* __restrict__ hin, float* __restrict__ hout, int ET) {
  long long t = (long long)blockIdx.x * 256 + threadIdx.x;
  if (t >= (long long)ET * HD) return;
  int e = (int)(t >> 4);
  int f = (int)(t & 15);
  int s = s32[e], d = d32[e];
  float a = ebuf[e] / (ssum[d] + 1e-16f);
  atomicAdd(hout + (size_t)d * HD + f, a * hin[(size_t)s * HD + f]);
}

// ---------------------------------------------------------------------------
// Final: softmax(h @ W2 + b2), 16 -> 40 per node. W2/b2 staged in LDS.
// ---------------------------------------------------------------------------
__global__ __launch_bounds__(256) void final_softmax(
    const float* __restrict__ h, const float* __restrict__ W2,
    const float* __restrict__ b2, float* __restrict__ out, int N) {
  __shared__ float w[HD * NCLASS];
  __shared__ float bb[NCLASS];
  for (int i = threadIdx.x; i < HD * NCLASS; i += 256) w[i] = W2[i];
  for (int i = threadIdx.x; i < NCLASS; i += 256) bb[i] = b2[i];
  __syncthreads();
  int n = blockIdx.x * 256 + threadIdx.x;
  if (n >= N) return;
  float hr[HD];
  const float4* hp = (const float4*)(h + (size_t)n * HD);
#pragma unroll
  for (int i = 0; i < 4; ++i) {
    float4 v = hp[i];
    hr[4*i+0] = v.x; hr[4*i+1] = v.y; hr[4*i+2] = v.z; hr[4*i+3] = v.w;
  }
  float logits[NCLASS];
  float mx = -3.402823466e+38f;
#pragma unroll 4
  for (int c = 0; c < NCLASS; ++c) {
    float acc = bb[c];
#pragma unroll
    for (int k = 0; k < HD; ++k) acc += hr[k] * w[k * NCLASS + c];
    logits[c] = acc;
    mx = fmaxf(mx, acc);
  }
  float sum = 0.f;
#pragma unroll 4
  for (int c = 0; c < NCLASS; ++c) {
    logits[c] = expf(logits[c] - mx);
    sum += logits[c];
  }
  float inv = 1.0f / sum;
  float* o = out + (size_t)n * NCLASS;
#pragma unroll 4
  for (int c = 0; c < NCLASS; ++c) o[c] = logits[c] * inv;
}

// ---------------------------------------------------------------------------
extern "C" void kernel_launch(void* const* d_in, const int* in_sizes, int n_in,
                              void* d_out, int out_size, void* d_ws, size_t ws_size,
                              hipStream_t stream) {
  const float*     x     = (const float*)d_in[0];
  const long long* ei    = (const long long*)d_in[1];
  const float*     W1    = (const float*)d_in[2];
  const float*     b1    = (const float*)d_in[3];
  const float*     beta1 = (const float*)d_in[4];
  const float*     beta2 = (const float*)d_in[5];
  const float*     beta3 = (const float*)d_in[6];
  const float*     W2    = (const float*)d_in[7];
  const float*     b2    = (const float*)d_in[8];
  float* out = (float*)d_out;

  const int N  = in_sizes[0] / FIN;     // 100000
  const int E  = in_sizes[1] / 2;       // 3200000
  const int ET = E + N;                 // + self-loops

  // Carve scratch (256 B aligned): ~60 MB total
  char* ws = (char*)d_ws;
  size_t off = 0;
  auto take = [&](size_t bytes) -> void* {
    void* p = (void*)(ws + off);
    off += (bytes + 255) & ~(size_t)255;
    return p;
  };
  float*    hA    = (float*)take((size_t)N * HD * 4);
  float*    hB    = (float*)take((size_t)N * HD * 4);
  float*    xn    = (float*)take((size_t)N * HD * 4);
  float*    albuf = (float*)take((size_t)ET * 4);
  unsigned* m_u   = (unsigned*)take((size_t)N * 4);
  float*    ssum  = (float*)take((size_t)N * 4);
  int*      s32   = (int*)take((size_t)ET * 4);
  int*      d32   = (int*)take((size_t)ET * 4);
  (void)ws_size; (void)n_in; (void)out_size;

  const int nbN  = (N + 255) / 256;
  const int nbE  = (ET + 255) / 256;
  const int nbA  = (int)(((long long)ET * HD + 255) / 256);
  const int numTiles = (N + 15) / 16;
  const int nbG  = (numTiles + 7) / 8;   // 8 waves (tiles) per 256-thread block

  build_edges<<<nbE, 256, 0, stream>>>(ei, s32, d32, E, ET);
  gemm1_relu<<<nbG, 256, 0, stream>>>(x, W1, b1, hA, N);

  auto conv = [&](const float* hin, float* hout, const float* beta) {
    rownorm16 <<<nbN, 256, 0, stream>>>(hin, xn, N);
    seg_init  <<<nbN, 256, 0, stream>>>(m_u, ssum, hout, N);
    edge_alpha<<<nbE, 256, 0, stream>>>(xn, s32, d32, beta, albuf, m_u, ET);
    edge_exp  <<<nbE, 256, 0, stream>>>(albuf, d32, m_u, ssum, ET);
    edge_aggr <<<nbA, 256, 0, stream>>>(albuf, s32, d32, ssum, hin, hout, ET);
  };
  conv(hA, hB, beta1);
  conv(hB, hA, beta2);
  conv(hA, hB, beta3);

  final_softmax<<<nbN, 256, 0, stream>>>(hB, W2, b2, out, N);
}
